// LSTMDecoder_47957604827203
// MI455X (gfx1250) — matrix operand order repaired and checked by reading, not compiled, not run on hardware
//
#include <hip/hip_runtime.h>

// ---------------------------------------------------------------------------
// LSTM decoder for MI455X (gfx1250): bf16 WMMA GEMMs with fused LSTM epilogue.
// B=2048, E=256, H=1024, V=42, SEQ=100.
// Round 3: launch_bounds(256,1) for full VGPR budget (no spills with 16 accs),
// unroll-by-2 K loops for cross-iteration load/WMMA pipelining, compile-time
// Kx via template (layer0 x-part fully unrolled).
// ---------------------------------------------------------------------------

#define B_    2048
#define E_    256
#define H_    1024
#define V_    42
#define VP_   48      // fc2 N padded to 3 WMMA tiles
#define XP_   64      // x (softmax feedback) K padded to 2 WMMA K-steps
#define G4H_  4096    // 4*H
#define T_    100
#define MT_   4       // M-tiles per wave in the LSTM cell kernel

typedef __attribute__((ext_vector_type(8)))  __bf16 v8bf;
typedef __attribute__((ext_vector_type(16))) __bf16 v16bf;
typedef __attribute__((ext_vector_type(8)))  float  v8f;

union FragU { v16bf v; v8bf h[2]; };

// A fragment: 16x32 bf16 tile, row-major source (leading dim `ld` elements).
// Lanes 0-15: M=0..15 with K {0..7, 16..23}; lanes 16-31: M=0..15 with
// K {8..15, 24..31}. Two contiguous 16B row segments per lane.
__device__ __forceinline__ v16bf load_a_frag(const __bf16* __restrict__ p0,
                                             int ld, int lane) {
  const int m  = lane & 15;
  const int hk = (lane >> 4) & 1;
  const __bf16* p = p0 + (size_t)m * ld + hk * 8;
  FragU f;
  f.h[0] = *(const v8bf*)(p);
  f.h[1] = *(const v8bf*)(p + 16);
  return f.v;
}

// B fragment: 32x16 bf16 tile == transpose of a 16(N)x32(K) row-major W block.
// Lane n holds column n; lanes 0-15 K {0..7, 16..23}, lanes 16-31 K {8..15,
// 24..31} -> two contiguous 16B row segments per lane.
__device__ __forceinline__ v16bf load_b_frag(const __bf16* __restrict__ p0,
                                             int ld, int lane) {
  const int n  = lane & 15;
  const int kh = (lane >> 4) & 1;
  const __bf16* p = p0 + (size_t)n * ld + kh * 8;
  FragU f;
  f.h[0] = *(const v8bf*)(p);
  f.h[1] = *(const v8bf*)(p + 16);
  return f.v;
}

__device__ __forceinline__ float sigm(float x) {
  return 1.0f / (1.0f + __expf(-x));
}
__device__ __forceinline__ float tanh_fast(float x) {
  float t = __expf(2.0f * x);           // saturates correctly at +/-inf
  return 1.0f - 2.0f / (t + 1.0f);
}

// ------------------------------- prep kernels ------------------------------

__global__ void k_f32_to_bf16(const float* __restrict__ s,
                              __bf16* __restrict__ d, int n) {
  int i = blockIdx.x * blockDim.x + threadIdx.x;
  if (i < n) d[i] = (__bf16)s[i];
}

__global__ void k_f32_to_bf16_pad(const float* __restrict__ s,
                                  __bf16* __restrict__ d,
                                  int sr, int sc, int dr, int dc) {
  int i = blockIdx.x * blockDim.x + threadIdx.x;
  if (i >= dr * dc) return;
  int r = i / dc, c = i % dc;
  d[i] = (r < sr && c < sc) ? (__bf16)s[r * sc + c] : (__bf16)0.0f;
}

__global__ void k_add_f32(const float* __restrict__ a,
                          const float* __restrict__ b,
                          float* __restrict__ o, int n) {
  int i = blockIdx.x * blockDim.x + threadIdx.x;
  if (i < n) o[i] = a[i] + b[i];
}

__global__ void k_pad_bias48(const float* __restrict__ b,
                             float* __restrict__ o) {
  int i = threadIdx.x;
  if (i < VP_) o[i] = (i < V_) ? b[i] : -1e30f;  // -1e30 masks softmax pad cols
}

__global__ void k_init_x(__bf16* __restrict__ x) {   // [B, XP_] one-hot at V-1
  int i = blockIdx.x * blockDim.x + threadIdx.x;
  if (i < B_ * XP_) {
    int c = i & (XP_ - 1);
    x[i] = (__bf16)((c == (V_ - 1)) ? 1.0f : 0.0f);
  }
}

// ------------------------------- fc1 GEMM ----------------------------------
// h0[2048,1024] = latent[2048,256] @ fc1_w^T + fc1_b   (output bf16, one-time)
__global__ __launch_bounds__(256)
void fc1_wmma(const __bf16* __restrict__ Lat,   // [B, E]
              const __bf16* __restrict__ W,     // [H, E]
              const float*  __restrict__ bias,  // [H] fp32
              __bf16* __restrict__ Hout) {      // [B, H]
  const int lane = threadIdx.x & 31;
  const int warp = threadIdx.x >> 5;
  const int n0 = (blockIdx.x * 8 + warp) * 16;
  const int m0 = blockIdx.y * 16;

  v8f acc = {};
#pragma unroll
  for (int k0 = 0; k0 < E_; k0 += 32) {
    v16bf a = load_a_frag(Lat + (size_t)m0 * E_ + k0, E_, lane);
    v16bf b = load_b_frag(W   + (size_t)n0 * E_ + k0, E_, lane);
    acc = __builtin_amdgcn_wmma_f32_16x16x32_bf16(false, a, false, b,
                                                  (short)0, acc, false, false);
  }
  const int col = n0 + (lane & 15);
  const float bv = bias[col];
  const int mb = m0 + ((lane >> 4) << 3);
#pragma unroll
  for (int r = 0; r < 8; ++r)
    Hout[(size_t)(mb + r) * H_ + col] = (__bf16)(acc[r] + bv);
}

// --------------------------- fused LSTM cell -------------------------------
// Each wave: 64 batch rows (4 M-tiles) x 16 hidden cols; 16 accumulators =
// i/f/g/o gates of the SAME 16 hidden units for 4 M-tiles. Per K-step:
// 4 A-frag + 4 B-frag loads feed 16 WMMAs (32 FLOP/byte from L2).
// Gate nonlinearity + cell update fused in registers.
// launch_bounds(256,1): one workgroup per unit -> full VGPR budget, no spills.
template <int KX>
__global__ __launch_bounds__(256, 1)
void lstm_cell_wmma(const __bf16* __restrict__ X,         // [B, KX]
                    const __bf16* __restrict__ Hprev,     // [B, H]
                    const __bf16* __restrict__ Wih,       // [4H, KX]
                    const __bf16* __restrict__ Whh,       // [4H, H]
                    const float*  __restrict__ bias,      // [4H] (ih+hh)
                    float* __restrict__ Cst,              // [B, H] inout
                    __bf16* __restrict__ Hout) {          // [B, H]
  const int lane = threadIdx.x & 31;
  const int warp = threadIdx.x >> 5;
  const int n0 = (blockIdx.x * 8 + warp) * 16;
  const int m0 = blockIdx.y * (MT_ * 16);

  v8f acc[MT_][4] = {};

  // x @ Wih^T contribution (KX=64 for layer0: fully unrolled, 2 steps)
#pragma unroll 2
  for (int k0 = 0; k0 < KX; k0 += 32) {
    v16bf af[MT_], bf[4];
#pragma unroll
    for (int mt = 0; mt < MT_; ++mt)
      af[mt] = load_a_frag(X + (size_t)(m0 + mt * 16) * KX + k0, KX, lane);
#pragma unroll
    for (int g = 0; g < 4; ++g)
      bf[g] = load_b_frag(Wih + (size_t)(g * H_ + n0) * KX + k0, KX, lane);
#pragma unroll
    for (int mt = 0; mt < MT_; ++mt)
#pragma unroll
      for (int g = 0; g < 4; ++g)
        acc[mt][g] = __builtin_amdgcn_wmma_f32_16x16x32_bf16(
            false, af[mt], false, bf[g], (short)0, acc[mt][g], false, false);
  }
  // h_prev @ Whh^T contribution
#pragma unroll 2
  for (int k0 = 0; k0 < H_; k0 += 32) {
    v16bf af[MT_], bf[4];
#pragma unroll
    for (int mt = 0; mt < MT_; ++mt)
      af[mt] = load_a_frag(Hprev + (size_t)(m0 + mt * 16) * H_ + k0, H_, lane);
#pragma unroll
    for (int g = 0; g < 4; ++g)
      bf[g] = load_b_frag(Whh + (size_t)(g * H_ + n0) * H_ + k0, H_, lane);
#pragma unroll
    for (int mt = 0; mt < MT_; ++mt)
#pragma unroll
      for (int g = 0; g < 4; ++g)
        acc[mt][g] = __builtin_amdgcn_wmma_f32_16x16x32_bf16(
            false, af[mt], false, bf[g], (short)0, acc[mt][g], false, false);
  }

  // Epilogue: torch gate order i,f,g,o. C-tile layout: VGPR r, lanes 0-15 ->
  // (M=r, N=lane); lanes 16-31 -> (M=8+r, N=lane-16).
  const int col = n0 + (lane & 15);
  const float bi = bias[0 * H_ + col];
  const float bf_ = bias[1 * H_ + col];
  const float bg = bias[2 * H_ + col];
  const float bo = bias[3 * H_ + col];
#pragma unroll
  for (int mt = 0; mt < MT_; ++mt) {
    const int mb = m0 + mt * 16 + ((lane >> 4) << 3);
#pragma unroll
    for (int r = 0; r < 8; ++r) {
      size_t idx = (size_t)(mb + r) * H_ + col;
      float iv = sigm(acc[mt][0][r] + bi);
      float fv = sigm(acc[mt][1][r] + bf_);
      float gv = tanh_fast(acc[mt][2][r] + bg);
      float ov = sigm(acc[mt][3][r] + bo);
      float c  = fv * Cst[idx] + iv * gv;
      Cst[idx] = c;
      Hout[idx] = (__bf16)(ov * tanh_fast(c));
    }
  }
}

// ------------------------- fc2 + softmax feedback --------------------------
// logits[B,42] = h1 @ fc2_w^T + b; logits -> d_out; softmax(bf16) -> xnext.
__global__ __launch_bounds__(256)
void fc2_softmax_wmma(const __bf16* __restrict__ H1,     // [B, H]
                      const __bf16* __restrict__ W,      // [VP_, H] (pad rows 0)
                      const float*  __restrict__ bias48, // [VP_] (pad = -1e30)
                      float* __restrict__ out, int t,
                      __bf16* __restrict__ xnext) {      // [B, XP_]
  const int lane = threadIdx.x & 31;
  const int warp = threadIdx.x >> 5;
  const int m0 = (blockIdx.x * 8 + warp) * 16;

  v8f acc[3] = {{}, {}, {}};
#pragma unroll 2
  for (int k0 = 0; k0 < H_; k0 += 32) {
    v16bf a = load_a_frag(H1 + (size_t)m0 * H_ + k0, H_, lane);
#pragma unroll
    for (int nt = 0; nt < 3; ++nt) {
      v16bf b = load_b_frag(W + (size_t)(nt * 16) * H_ + k0, H_, lane);
      acc[nt] = __builtin_amdgcn_wmma_f32_16x16x32_bf16(false, a, false, b,
                                                        (short)0, acc[nt],
                                                        false, false);
    }
  }
  const int nl = lane & 15;
  const int mb = m0 + ((lane >> 4) << 3);
#pragma unroll
  for (int r = 0; r < 8; ++r) {
    float l0 = acc[0][r] + bias48[nl];
    float l1 = acc[1][r] + bias48[16 + nl];
    float l2 = acc[2][r] + bias48[32 + nl];   // cols 42-47: bias -1e30 -> masked
    const int b = mb + r;
    size_t ob = (size_t)b * (T_ * V_) + (size_t)t * V_;
    out[ob + nl] = l0;
    out[ob + 16 + nl] = l1;
    if (32 + nl < V_) out[ob + 32 + nl] = l2;

    // row softmax: each row lives in one 16-lane group (3 cols per lane)
    float mx = fmaxf(fmaxf(l0, l1), l2);
    for (int s = 1; s < 16; s <<= 1) mx = fmaxf(mx, __shfl_xor(mx, s, 32));
    float e0 = __expf(l0 - mx), e1 = __expf(l1 - mx), e2 = __expf(l2 - mx);
    float sm = e0 + e1 + e2;
    for (int s = 1; s < 16; s <<= 1) sm += __shfl_xor(sm, s, 32);
    float inv = 1.0f / sm;
    __bf16* xr = xnext + (size_t)b * XP_;
    xr[nl]      = (__bf16)(e0 * inv);
    xr[16 + nl] = (__bf16)(e1 * inv);
    xr[32 + nl] = (__bf16)((32 + nl < V_) ? (e2 * inv) : 0.0f);
  }
}

// ------------------------------- host glue ---------------------------------

static inline int cdiv(int a, int b) { return (a + b - 1) / b; }

extern "C" void kernel_launch(void* const* d_in, const int* in_sizes, int n_in,
                              void* d_out, int out_size, void* d_ws,
                              size_t ws_size, hipStream_t stream) {
  const float* latent = (const float*)d_in[0];
  const float* fc1_w  = (const float*)d_in[1];
  const float* fc1_b  = (const float*)d_in[2];
  const float* fc2_w  = (const float*)d_in[3];
  const float* fc2_b  = (const float*)d_in[4];
  const float* w_ih0  = (const float*)d_in[5];
  const float* w_hh0  = (const float*)d_in[6];
  const float* b_ih0  = (const float*)d_in[7];
  const float* b_hh0  = (const float*)d_in[8];
  const float* w_ih1  = (const float*)d_in[9];
  const float* w_hh1  = (const float*)d_in[10];
  const float* b_ih1  = (const float*)d_in[11];
  const float* b_hh1  = (const float*)d_in[12];
  (void)in_sizes; (void)n_in; (void)out_size; (void)ws_size;

  uint8_t* wsp = (uint8_t*)d_ws;
  auto alloc = [&](size_t bytes) -> void* {
    void* p = (void*)wsp;
    wsp += (bytes + 255) & ~(size_t)255;
    return p;
  };

  __bf16* wih0p = (__bf16*)alloc((size_t)G4H_ * XP_ * 2);  // padded K 42->64
  __bf16* whh0b = (__bf16*)alloc((size_t)G4H_ * H_ * 2);
  __bf16* wih1b = (__bf16*)alloc((size_t)G4H_ * H_ * 2);
  __bf16* whh1b = (__bf16*)alloc((size_t)G4H_ * H_ * 2);
  __bf16* fc1wb = (__bf16*)alloc((size_t)H_ * E_ * 2);
  __bf16* fc2wb = (__bf16*)alloc((size_t)VP_ * H_ * 2);    // padded N 42->48
  float*  bias0 = (float*)alloc((size_t)G4H_ * 4);
  float*  bias1 = (float*)alloc((size_t)G4H_ * 4);
  float*  fc2b48 = (float*)alloc((size_t)VP_ * 4);
  __bf16* latb  = (__bf16*)alloc((size_t)B_ * E_ * 2);
  __bf16* h0A   = (__bf16*)alloc((size_t)B_ * H_ * 2);
  __bf16* h0B   = (__bf16*)alloc((size_t)B_ * H_ * 2);
  __bf16* h1A   = (__bf16*)alloc((size_t)B_ * H_ * 2);
  __bf16* h1B   = (__bf16*)alloc((size_t)B_ * H_ * 2);
  float*  c0    = (float*)alloc((size_t)B_ * H_ * 4);
  float*  c1    = (float*)alloc((size_t)B_ * H_ * 4);
  __bf16* xpad  = (__bf16*)alloc((size_t)B_ * XP_ * 2);

  const int TB = 256;

  // ---- one-time prep (runs every launch; deterministic) ----
  k_f32_to_bf16_pad<<<cdiv(G4H_ * XP_, TB), TB, 0, stream>>>(
      w_ih0, wih0p, G4H_, V_, G4H_, XP_);
  k_f32_to_bf16<<<cdiv(G4H_ * H_, TB), TB, 0, stream>>>(w_hh0, whh0b, G4H_ * H_);
  k_f32_to_bf16<<<cdiv(G4H_ * H_, TB), TB, 0, stream>>>(w_ih1, wih1b, G4H_ * H_);
  k_f32_to_bf16<<<cdiv(G4H_ * H_, TB), TB, 0, stream>>>(w_hh1, whh1b, G4H_ * H_);
  k_f32_to_bf16<<<cdiv(H_ * E_, TB), TB, 0, stream>>>(fc1_w, fc1wb, H_ * E_);
  k_f32_to_bf16_pad<<<cdiv(VP_ * H_, TB), TB, 0, stream>>>(
      fc2_w, fc2wb, V_, H_, VP_, H_);
  k_add_f32<<<cdiv(G4H_, TB), TB, 0, stream>>>(b_ih0, b_hh0, bias0, G4H_);
  k_add_f32<<<cdiv(G4H_, TB), TB, 0, stream>>>(b_ih1, b_hh1, bias1, G4H_);
  k_pad_bias48<<<1, 64, 0, stream>>>(fc2_b, fc2b48);
  k_f32_to_bf16<<<cdiv(B_ * E_, TB), TB, 0, stream>>>(latent, latb, B_ * E_);
  k_init_x<<<cdiv(B_ * XP_, TB), TB, 0, stream>>>(xpad);
  hipMemsetAsync(c0, 0, (size_t)B_ * H_ * 4, stream);
  hipMemsetAsync(c1, 0, (size_t)B_ * H_ * 4, stream);
  hipMemsetAsync(h1A, 0, (size_t)B_ * H_ * 2, stream);

  // ---- h0 = fc1(latent) ----
  fc1_wmma<<<dim3(H_ / 16 / 8, B_ / 16), TB, 0, stream>>>(latb, fc1wb, fc1_b, h0A);

  // ---- 100 recurrent steps ----
  __bf16 *h0c = h0A, *h0n = h0B, *h1c = h1A, *h1n = h1B;
  const dim3 gcell(H_ / 16 / 8, B_ / (MT_ * 16));  // 8 waves/block; 4 M-tiles/wave
  for (int t = 0; t < T_; ++t) {
    lstm_cell_wmma<XP_><<<gcell, TB, 0, stream>>>(xpad, h0c, wih0p, whh0b,
                                                  bias0, c0, h0n);
    lstm_cell_wmma<H_><<<gcell, TB, 0, stream>>>(h0n, h1c, wih1b, whh1b,
                                                 bias1, c1, h1n);
    fc2_softmax_wmma<<<B_ / 16 / 8, TB, 0, stream>>>(h1n, fc2wb, fc2b48,
                                                     (float*)d_out, t, xpad);
    __bf16* tmp;
    tmp = h0c; h0c = h0n; h0n = tmp;
    tmp = h1c; h1c = h1n; h1n = tmp;
  }
}
